// WaveNet_87780541596472
// MI455X (gfx1250) — matrix operand order repaired and checked by reading, not compiled, not run on hardware
//
#include <hip/hip_runtime.h>
#include <math.h>

// WaveNet on gfx1250: fused per-block WMMA kernels (v_wmma_f32_16x16x32_f16),
// weight staging via CDNA5 async global->LDS DMA (ASYNCcnt path).
// B=64, CIN=8, L=4096, RC=64, SC=128, K=2, 30 blocks, dilations 2^(i%10).

typedef __attribute__((ext_vector_type(16))) _Float16 v16h;
typedef __attribute__((ext_vector_type(8)))  _Float16 v8h;
typedef __attribute__((ext_vector_type(8)))  float    v8f;

union AFrag { v16h v; v8h h8[2]; };
union Pack8 { v8h v; _Float16 h[8]; };

#define Bsz 64
#define RC  64
#define SC  128
#define Lsz 4096
#define NBLK 30

// ---------------------------------------------------------------------------
// CDNA5 async global->LDS copy helpers (not covered by builtins).
// LDS flat addresses: low 32 bits == LDS byte offset (ISA aperture rule).
// ---------------------------------------------------------------------------
__device__ __forceinline__ unsigned lds_off(const void* p) {
  return (unsigned)(unsigned long long)p;
}
__device__ __forceinline__ void async_ld16(unsigned dst_lds, const void* src) {
  asm volatile("global_load_async_to_lds_b128 %0, %1, off"
               :: "v"(dst_lds), "v"(src));
}
__device__ __forceinline__ void wait_async0() {
  asm volatile("s_wait_asynccnt 0x0" ::: "memory");
}

// ---------------------------------------------------------------------------
// Workspace layout (bytes). Requires ws_size >= ~258 MB.
// ---------------------------------------------------------------------------
static const size_t OFF_HA     = 0;                         // 64 MB fp32 h ping
static const size_t OFF_HB     = 67108864ull;               // 64 MB fp32 h pong
static const size_t OFF_SKIP   = 134217728ull;              // 128 MB fp32 skip
static const size_t OFF_POOLED = 268435456ull;              // 32 KB fp32 pooled sums
static const size_t OFF_WDIL   = 268468224ull;              // 30*128*128 f16
static const size_t OFF_ARS    = 269451264ull;              // 30*192*64 f16
static const size_t OFF_W1     = 270188544ull;              // 128*128 f16
static const size_t OFF_W2     = 270221312ull;              // 128*128 f16

// ---------------------------------------------------------------------------
// Small utility kernels
// ---------------------------------------------------------------------------
__global__ void zero_f32(float* __restrict__ p, int n) {
  for (int i = blockIdx.x * blockDim.x + threadIdx.x; i < n; i += gridDim.x * blockDim.x)
    p[i] = 0.0f;
}

// w_dil (30,128,64,2) fp32 -> (30,128,128) f16, K = [tap0 ch0..63 | tap1 ch0..63]
__global__ void cvt_dil(const float* __restrict__ w_dil, _Float16* __restrict__ dst) {
  int idx = blockIdx.x * blockDim.x + threadIdx.x;
  if (idx >= NBLK * 128 * 128) return;
  int i = idx >> 14, rem = idx & 16383;
  int o = rem >> 7, k = rem & 127;
  float v = (k < 64) ? w_dil[((size_t)(i * 128 + o) * 64 + k) * 2 + 0]
                     : w_dil[((size_t)(i * 128 + o) * 64 + (k - 64)) * 2 + 1];
  dst[idx] = (_Float16)v;
}

// [w_res (64x64); w_skip (128x64)] -> (30,192,64) f16
__global__ void cvt_rs(const float* __restrict__ w_res, const float* __restrict__ w_skip,
                       _Float16* __restrict__ dst) {
  int idx = blockIdx.x * blockDim.x + threadIdx.x;
  if (idx >= NBLK * 192 * 64) return;
  int i = idx / (192 * 64), rem = idx % (192 * 64);
  int m = rem >> 6, c = rem & 63;
  float v = (m < 64) ? w_res[(size_t)(i * 64 + m) * 64 + c]
                     : w_skip[(size_t)(i * 128 + (m - 64)) * 64 + c];
  dst[idx] = (_Float16)v;
}

__global__ void cvt_f16(const float* __restrict__ src, _Float16* __restrict__ dst, int n) {
  for (int i = blockIdx.x * blockDim.x + threadIdx.x; i < n; i += gridDim.x * blockDim.x)
    dst[i] = (_Float16)src[i];
}

// h0[b,o,l] = sum_c w_in[o,c]*x[b,c,l] + b_in[o]   (CIN=8, bandwidth bound)
__global__ void inproj_kernel(const float* __restrict__ x, const float* __restrict__ w_in,
                              const float* __restrict__ b_in, float* __restrict__ h) {
  int idx = blockIdx.x * blockDim.x + threadIdx.x; // B*RC*L = 16.7M
  int l = idx & (Lsz - 1);
  int o = (idx >> 12) & 63;
  int b = idx >> 18;
  float s = b_in[o];
#pragma unroll
  for (int c = 0; c < 8; ++c)
    s += w_in[o * 8 + c] * x[((size_t)b * 8 + c) * Lsz + l];
  h[idx] = s;
}

// ---------------------------------------------------------------------------
// Fused residual block: dilated conv (128x128 GEMM) -> gate -> [res;skip] GEMM
// One workgroup = (batch b, 64 positions). 8 waves, LDS-resident f16 weights.
// ---------------------------------------------------------------------------
struct __align__(32) BlockSmem {
  _Float16 Adil[128][128];  // 32 KB  conv weights, K-concatenated taps
  _Float16 Ars[192][64];    // 24 KB  [w_res; w_skip]
  _Float16 Bt[64][128];     // 16 KB  activations, [n][k] (k = prev|cur channels)
  _Float16 Gt[64][64];      //  8 KB  gated, [n][c]
  float    Hc[64][64];      // 16 KB  fp32 cur h tile (residual add)
  float    Cf[128][64];     // 32 KB  conv_out fp32
};

__global__ __launch_bounds__(256) void wavenet_block_kernel(
    const float* __restrict__ h_in, float* __restrict__ h_out, float* __restrict__ skip,
    const _Float16* __restrict__ wdil16, const _Float16* __restrict__ ars16,
    const float* __restrict__ b_dil, const float* __restrict__ b_res,
    const float* __restrict__ b_skip, int dilation) {
  __shared__ BlockSmem sm;
  const int tid  = threadIdx.x;
  const int wv   = tid >> 5, lane = tid & 31;
  const int lr   = lane & 15, grp = lane >> 4;
  const int b    = blockIdx.x >> 6;
  const int l0   = (blockIdx.x & 63) * 64;
  const float* hb = h_in + (size_t)b * RC * Lsz;

  // Async-DMA the f16 weights straight into LDS (no VGPR round-trip);
  // overlaps with the VALU-heavy activation staging below.
  {
    unsigned a0 = lds_off(&sm.Adil[0][0]);
    const char* g0 = (const char*)wdil16;
    for (int i = tid; i < 2048; i += 256)          // 2048 x 16B = 32 KB
      async_ld16(a0 + i * 16, g0 + (size_t)i * 16);
    unsigned a1 = lds_off(&sm.Ars[0][0]);
    const char* g1 = (const char*)ars16;
    for (int i = tid; i < 1536; i += 256)          // 1536 x 16B = 24 KB
      async_ld16(a1 + i * 16, g1 + (size_t)i * 16);
  }
  // Stage activations: Bt[n][0..63] = h[l-d] (tap0), Bt[n][64..127] = h[l] (tap1)
  for (int t = tid; t < 64 * 64; t += 256) {
    int c = t >> 6, n = t & 63;
    float cur = hb[(size_t)c * Lsz + l0 + n];
    sm.Hc[c][n] = cur;
    sm.Bt[n][64 + c] = (_Float16)cur;
    int lp = l0 + n - dilation;
    float pv = (lp >= 0) ? hb[(size_t)c * Lsz + lp] : 0.0f;
    sm.Bt[n][c] = (_Float16)pv;
  }
  wait_async0();
  __syncthreads();

  // GEMM1: conv_out(128 x 64) = Adil(128x128) x Bt^T(128x64)
  {
    const int mt = wv;
    const int row = mt * 16 + lr;
#pragma unroll
    for (int nt = 0; nt < 4; ++nt) {
      const int col = nt * 16 + lr;
      v8f acc = {};
#pragma unroll
      for (int ks = 0; ks < 4; ++ks) {
        AFrag a;
        int kb = ks * 32 + grp * 16;
        a.h8[0] = *(const v8h*)&sm.Adil[row][kb];
        a.h8[1] = *(const v8h*)&sm.Adil[row][kb + 8];
        v16h bf = *(const v16h*)&sm.Bt[col][ks * 32 + grp * 16];
        acc = __builtin_amdgcn_wmma_f32_16x16x32_f16(false, a.v, false, bf,
                                                     (short)0, acc, false, false);
      }
#pragma unroll
      for (int r = 0; r < 8; ++r)
        sm.Cf[mt * 16 + grp * 8 + r][col] = acc[r];
    }
  }
  __syncthreads();

  // Gate: gated = tanh(t + bt) * sigmoid(s + bs), store transposed f16
  for (int t = tid; t < 64 * 64; t += 256) {
    int c = t >> 6, n = t & 63;
    float tv = sm.Cf[c][n] + b_dil[c];
    float sv = sm.Cf[64 + c][n] + b_dil[64 + c];
    float g = tanhf(tv) * (1.0f / (1.0f + __expf(-sv)));
    sm.Gt[n][c] = (_Float16)g;
  }
  __syncthreads();

  // GEMM2: [h_res; skip](192 x 64) = Ars(192x64) x Gt^T(64x64), fused writeback
  {
    float* houtb = h_out + (size_t)b * RC * Lsz;
    float* skipb = skip + (size_t)b * SC * Lsz;
    for (int job = wv; job < 48; job += 8) {   // 12 M-tiles x 4 N-tiles
      const int mt = job >> 2, nt = job & 3;
      const int row = mt * 16 + lr;
      const int col = nt * 16 + lr;
      v8f acc = {};
#pragma unroll
      for (int ks = 0; ks < 2; ++ks) {
        AFrag a;
        int kb = ks * 32 + grp * 16;
        a.h8[0] = *(const v8h*)&sm.Ars[row][kb];
        a.h8[1] = *(const v8h*)&sm.Ars[row][kb + 8];
        v16h bf = *(const v16h*)&sm.Gt[col][ks * 32 + grp * 16];
        acc = __builtin_amdgcn_wmma_f32_16x16x32_f16(false, a.v, false, bf,
                                                     (short)0, acc, false, false);
      }
#pragma unroll
      for (int r = 0; r < 8; ++r) {
        int M = mt * 16 + grp * 8 + r;
        if (M < 64) {
          houtb[(size_t)M * Lsz + l0 + col] = acc[r] + b_res[M] + sm.Hc[M][col];
        } else {
          int sc = M - 64;
          float* p = &skipb[(size_t)sc * Lsz + l0 + col];
          *p = *p + acc[r] + b_skip[sc];
        }
      }
    }
  }
}

// ---------------------------------------------------------------------------
// Output head: relu(skip) -> w_out1 -> relu -> w_out2 -> partial mean (atomic)
// ---------------------------------------------------------------------------
struct __align__(32) OutSmem {
  _Float16 W1[128][128];   // 32 KB
  _Float16 W2[128][128];   // 32 KB
  _Float16 Bt2[64][128];   // 16 KB  relu(skip) tile, [n][c]
  _Float16 Bt3[64][128];   // 16 KB  relu(out1) tile, [n][c]
  float    Cf[128][64];    // 32 KB
};

__global__ __launch_bounds__(256) void wavenet_out_kernel(
    const float* __restrict__ skip, const _Float16* __restrict__ w1_16,
    const _Float16* __restrict__ w2_16, const float* __restrict__ b1,
    const float* __restrict__ b2, float* __restrict__ pooled) {
  __shared__ OutSmem sm;
  const int tid = threadIdx.x;
  const int wv = tid >> 5, lane = tid & 31;
  const int lr = lane & 15, grp = lane >> 4;
  const int b = blockIdx.x >> 6;
  const int l0 = (blockIdx.x & 63) * 64;
  const float* skipb = skip + (size_t)b * SC * Lsz;

  {
    unsigned a0 = lds_off(&sm.W1[0][0]);
    const char* g0 = (const char*)w1_16;
    for (int i = tid; i < 2048; i += 256)
      async_ld16(a0 + i * 16, g0 + (size_t)i * 16);
    unsigned a1 = lds_off(&sm.W2[0][0]);
    const char* g1 = (const char*)w2_16;
    for (int i = tid; i < 2048; i += 256)
      async_ld16(a1 + i * 16, g1 + (size_t)i * 16);
  }
  for (int t = tid; t < 128 * 64; t += 256) {
    int c = t >> 6, n = t & 63;
    sm.Bt2[n][c] = (_Float16)fmaxf(skipb[(size_t)c * Lsz + l0 + n], 0.0f);
  }
  wait_async0();
  __syncthreads();

  // GEMM1: out1 = relu(W1 @ rskip + b1) -> Bt3 (f16, transposed)
  {
    const int mt = wv;
    const int row = mt * 16 + lr;
#pragma unroll
    for (int nt = 0; nt < 4; ++nt) {
      const int col = nt * 16 + lr;
      v8f acc = {};
#pragma unroll
      for (int ks = 0; ks < 4; ++ks) {
        AFrag a;
        int kb = ks * 32 + grp * 16;
        a.h8[0] = *(const v8h*)&sm.W1[row][kb];
        a.h8[1] = *(const v8h*)&sm.W1[row][kb + 8];
        v16h bf = *(const v16h*)&sm.Bt2[col][ks * 32 + grp * 16];
        acc = __builtin_amdgcn_wmma_f32_16x16x32_f16(false, a.v, false, bf,
                                                     (short)0, acc, false, false);
      }
      Pack8 pk;
#pragma unroll
      for (int r = 0; r < 8; ++r) {
        int M = mt * 16 + grp * 8 + r;
        pk.h[r] = (_Float16)fmaxf(acc[r] + b1[M], 0.0f);
      }
      *(v8h*)&sm.Bt3[col][mt * 16 + grp * 8] = pk.v;
    }
  }
  __syncthreads();

  // GEMM2: out2 = W2 @ out1 + b2 -> Cf
  {
    const int mt = wv;
    const int row = mt * 16 + lr;
#pragma unroll
    for (int nt = 0; nt < 4; ++nt) {
      const int col = nt * 16 + lr;
      v8f acc = {};
#pragma unroll
      for (int ks = 0; ks < 4; ++ks) {
        AFrag a;
        int kb = ks * 32 + grp * 16;
        a.h8[0] = *(const v8h*)&sm.W2[row][kb];
        a.h8[1] = *(const v8h*)&sm.W2[row][kb + 8];
        v16h bf = *(const v16h*)&sm.Bt3[col][ks * 32 + grp * 16];
        acc = __builtin_amdgcn_wmma_f32_16x16x32_f16(false, a.v, false, bf,
                                                     (short)0, acc, false, false);
      }
#pragma unroll
      for (int r = 0; r < 8; ++r) {
        int M = mt * 16 + grp * 8 + r;
        sm.Cf[M][col] = acc[r] + b2[M];
      }
    }
  }
  __syncthreads();

  // Partial sum over the 64 positions -> pooled accumulator
  if (tid < 128) {
    float s = 0.0f;
#pragma unroll 8
    for (int n = 0; n < 64; ++n) s += sm.Cf[tid][n];
    atomicAdd(&pooled[b * SC + tid], s);
  }
}

// ---------------------------------------------------------------------------
// FC head (tiny): mean -> fc1(256) relu -> fc2(128) relu -> fc3(6)
// ---------------------------------------------------------------------------
__global__ __launch_bounds__(256) void wavenet_head_kernel(
    const float* __restrict__ pooled, const float* __restrict__ w_fc1,
    const float* __restrict__ b_fc1, const float* __restrict__ w_fc2,
    const float* __restrict__ b_fc2, const float* __restrict__ w_fc3,
    const float* __restrict__ b_fc3, float* __restrict__ out) {
  __shared__ float P[64][128];
  __shared__ float Z1[64][256];
  __shared__ float Z2[64][128];
  const int tid = threadIdx.x;
  const float inv_l = 1.0f / (float)Lsz;
  for (int t = tid; t < 64 * 128; t += 256)
    P[t >> 7][t & 127] = pooled[t] * inv_l;
  __syncthreads();
  for (int t = tid; t < 64 * 256; t += 256) {
    int bi = t >> 8, o = t & 255;
    float s = b_fc1[o];
    for (int c = 0; c < 128; ++c) s += P[bi][c] * w_fc1[o * 128 + c];
    Z1[bi][o] = fmaxf(s, 0.0f);
  }
  __syncthreads();
  for (int t = tid; t < 64 * 128; t += 256) {
    int bi = t >> 7, o = t & 127;
    float s = b_fc2[o];
    for (int c = 0; c < 256; ++c) s += Z1[bi][c] * w_fc2[o * 256 + c];
    Z2[bi][o] = fmaxf(s, 0.0f);
  }
  __syncthreads();
  if (tid < 64 * 6) {
    int bi = tid / 6, o = tid % 6;
    float s = b_fc3[o];
    for (int c = 0; c < 128; ++c) s += Z2[bi][c] * w_fc3[o * 128 + c];
    out[bi * 6 + o] = s;
  }
}

// ---------------------------------------------------------------------------
extern "C" void kernel_launch(void* const* d_in, const int* in_sizes, int n_in,
                              void* d_out, int out_size, void* d_ws, size_t ws_size,
                              hipStream_t stream) {
  const float* x      = (const float*)d_in[0];
  const float* w_in   = (const float*)d_in[1];
  const float* b_in   = (const float*)d_in[2];
  const float* w_dil  = (const float*)d_in[3];
  const float* b_dil  = (const float*)d_in[4];
  const float* w_res  = (const float*)d_in[5];
  const float* b_res  = (const float*)d_in[6];
  const float* w_skip = (const float*)d_in[7];
  const float* b_skip = (const float*)d_in[8];
  const float* w_out1 = (const float*)d_in[9];
  const float* b_out1 = (const float*)d_in[10];
  const float* w_out2 = (const float*)d_in[11];
  const float* b_out2 = (const float*)d_in[12];
  const float* w_fc1  = (const float*)d_in[13];
  const float* b_fc1  = (const float*)d_in[14];
  const float* w_fc2  = (const float*)d_in[15];
  const float* b_fc2  = (const float*)d_in[16];
  const float* w_fc3  = (const float*)d_in[17];
  const float* b_fc3  = (const float*)d_in[18];

  char* ws = (char*)d_ws;
  float*    hA     = (float*)(ws + OFF_HA);
  float*    hB     = (float*)(ws + OFF_HB);
  float*    skip   = (float*)(ws + OFF_SKIP);
  float*    pooled = (float*)(ws + OFF_POOLED);
  _Float16* wdil16 = (_Float16*)(ws + OFF_WDIL);
  _Float16* ars16  = (_Float16*)(ws + OFF_ARS);
  _Float16* w1_16  = (_Float16*)(ws + OFF_W1);
  _Float16* w2_16  = (_Float16*)(ws + OFF_W2);

  // Weight conversion / repacking (f32 -> f16, conv taps K-concatenated)
  cvt_dil<<<(NBLK * 128 * 128 + 255) / 256, 256, 0, stream>>>(w_dil, wdil16);
  cvt_rs<<<(NBLK * 192 * 64 + 255) / 256, 256, 0, stream>>>(w_res, w_skip, ars16);
  cvt_f16<<<64, 256, 0, stream>>>(w_out1, w1_16, SC * SC);
  cvt_f16<<<64, 256, 0, stream>>>(w_out2, w2_16, SC * SC);

  // Zero skip accumulator + pooled (contiguous region)
  zero_f32<<<4096, 256, 0, stream>>>(skip, Bsz * SC * Lsz + Bsz * SC);

  // Input projection
  inproj_kernel<<<(Bsz * RC * Lsz) / 256, 256, 0, stream>>>(x, w_in, b_in, hA);

  // 30 fused residual blocks (ping-pong h)
  const dim3 grid(Bsz * (Lsz / 64));
  for (int i = 0; i < NBLK; ++i) {
    const float* src = (i & 1) ? hB : hA;
    float*       dst = (i & 1) ? hA : hB;
    int dil = 1 << (i % 10);
    wavenet_block_kernel<<<grid, 256, 0, stream>>>(
        src, dst, skip,
        wdil16 + (size_t)i * 128 * 128, ars16 + (size_t)i * 192 * 64,
        b_dil + (size_t)i * 128, b_res + (size_t)i * 64, b_skip + (size_t)i * 128, dil);
  }

  // Output 1x1 convs + pooling
  wavenet_out_kernel<<<grid, 256, 0, stream>>>(skip, w1_16, w2_16, b_out1, b_out2, pooled);

  // FC head
  wavenet_head_kernel<<<1, 256, 0, stream>>>(pooled, w_fc1, b_fc1, w_fc2, b_fc2,
                                             w_fc3, b_fc3, (float*)d_out);
}
